// MultiHeadSelfAttention_withLoop_50268297232999
// MI455X (gfx1250) — matrix operand order repaired and checked by reading, not compiled, not run on hardware
//
#include <hip/hip_runtime.h>

#define B_  16
#define S_  1024
#define D_  768
#define H_  12
#define DH  64
#define BH  (B_ * H_)

typedef __attribute__((ext_vector_type(16))) __bf16 v16bf;
typedef __attribute__((ext_vector_type(8)))  __bf16 v8bf;
typedef __attribute__((ext_vector_type(4)))  __bf16 v4bf;
typedef __attribute__((ext_vector_type(8)))  float  v8f;
typedef __attribute__((ext_vector_type(4)))  float  v4f;

#if __has_builtin(__builtin_amdgcn_global_load_async_to_lds_b128)
#define HAS_ASYNC 1
// exact parameter types per compiler diagnostic: AS1/AS3 pointers to int4
typedef int v4i_vs __attribute__((vector_size(16)));
typedef __attribute__((address_space(1))) v4i_vs* as1_v4i_p;
typedef __attribute__((address_space(3))) v4i_vs* as3_v4i_p;
#endif

static __device__ __forceinline__ __bf16 tobf(float f) { return (__bf16)f; }

static __device__ __forceinline__ v16bf cat16(v8bf lo, v8bf hi) {
  v16bf r;
#pragma unroll
  for (int i = 0; i < 8; ++i) { r[i] = lo[i]; r[i + 8] = hi[i]; }
  return r;
}

// 16B global -> LDS copy. Async (ASYNCcnt) on gfx1250, sync fallback otherwise.
static __device__ __forceinline__ void cp16(const __bf16* g, __bf16* l) {
#ifdef HAS_ASYNC
  __builtin_amdgcn_global_load_async_to_lds_b128(
      (as1_v4i_p)(unsigned long long)(const void*)g,
      (as3_v4i_p)(unsigned)(unsigned long long)(const void*)l,
      0, 0);
#else
  *(v8bf*)l = *(const v8bf*)g;
#endif
}

static __device__ __forceinline__ void async_wait0() {
#ifdef HAS_ASYNC
#if __has_builtin(__builtin_amdgcn_s_wait_asynccnt)
  __builtin_amdgcn_s_wait_asynccnt(0);
#else
  asm volatile("s_wait_asynccnt 0x0" ::: "memory");
#endif
#endif
}

// xor-swizzle reductions within each 16-lane group (rows of a WMMA C tile)
static __device__ __forceinline__ float redmax16(float v) {
  v = fmaxf(v, __int_as_float(__builtin_amdgcn_ds_swizzle(__float_as_int(v), 0x041F)));
  v = fmaxf(v, __int_as_float(__builtin_amdgcn_ds_swizzle(__float_as_int(v), 0x081F)));
  v = fmaxf(v, __int_as_float(__builtin_amdgcn_ds_swizzle(__float_as_int(v), 0x101F)));
  v = fmaxf(v, __int_as_float(__builtin_amdgcn_ds_swizzle(__float_as_int(v), 0x201F)));
  return v;
}
static __device__ __forceinline__ float redsum16(float v) {
  v += __int_as_float(__builtin_amdgcn_ds_swizzle(__float_as_int(v), 0x041F));
  v += __int_as_float(__builtin_amdgcn_ds_swizzle(__float_as_int(v), 0x081F));
  v += __int_as_float(__builtin_amdgcn_ds_swizzle(__float_as_int(v), 0x101F));
  v += __int_as_float(__builtin_amdgcn_ds_swizzle(__float_as_int(v), 0x201F));
  return v;
}

// one 16x16 projection tile: x[64x64] @ W^T[64x64], K=64 in two 32-chunks
static __device__ __forceinline__ v8f proj_tile(
    const __bf16 (*x)[72], const __bf16 (*wT)[72], int mi, int ni, int lane) {
  const int lh = lane >> 4, ln = lane & 15;
  const int m = mi * 16 + ln;
  v8f acc = {};
#pragma unroll
  for (int kc = 0; kc < 64; kc += 32) {
    v16bf a = cat16(*(const v8bf*)&x[m][kc + 8 * lh],
                    *(const v8bf*)&x[m][kc + 16 + 8 * lh]);
    v16bf bm;
#pragma unroll
    for (int s = 0; s < 16; ++s) bm[s] = wT[kc + (s & 7) + 8 * lh + 16 * (s >> 3)][ni * 16 + ln];
    acc = __builtin_amdgcn_wmma_f32_16x16x32_bf16(
        false, a, false, bm, (short)0, acc, false, false);
  }
  return acc;
}

// ---------------------------------------------------------------------------
// Kernel 1: fused QKV projection. One block = (b,h, 64-row s tile).
// Workspace layouts (bf16): Q row-major [s][e], K transposed [e][s],
// V row-major [s][e] -- chosen so the attention kernel's staging copies are
// all straight (transpose-free) and therefore async-LDS friendly.
// Softmax scale 1/sqrt(Dh) folded into Q.
// ---------------------------------------------------------------------------
__global__ __launch_bounds__(256) void qkv_proj_wmma(
    const float* __restrict__ tokens,
    const float* __restrict__ Wq, const float* __restrict__ bq,
    const float* __restrict__ Wk, const float* __restrict__ bk,
    const float* __restrict__ Wv, const float* __restrict__ bv,
    __bf16* __restrict__ qkvT)
{
  __shared__ __align__(16) __bf16 x_lds[64][72];        // x tile [s][d]
  __shared__ __align__(16) __bf16 wT_lds[3][64][72];    // W^T   [d][e]
  __shared__ __align__(16) __bf16 o_lds[64][72];        // out restage [s][e]

  const int tid = threadIdx.x;
  const int bh  = blockIdx.x;
  const int b   = bh / H_;
  const int h   = bh % H_;
  const int s0  = blockIdx.y * 64;

  // stage x tile (64x64 fp32 -> bf16), coalesced 16B loads
#pragma unroll
  for (int c = 0; c < 4; ++c) {
    int j  = c * 256 + tid;                 // 1024 float4 units
    int r  = j >> 4, c4 = j & 15;
    v4f xv = *(const v4f*)(tokens + (long)(b * S_ + s0 + r) * D_ + h * DH + c4 * 4);
    v4bf xb;
#pragma unroll
    for (int i = 0; i < 4; ++i) xb[i] = tobf(xv[i]);
    *(v4bf*)&x_lds[r][c4 * 4] = xb;
  }

  // stage Wq/Wk/Wv transposed into LDS (bf16)
#pragma unroll
  for (int p = 0; p < 12; ++p) {
    const int mat = p >> 2;
    int rem = (p & 3) * 256 + tid;          // 0..1023
    int e = rem >> 4, d4 = rem & 15;
    const float* Wm = (mat == 0 ? Wq : (mat == 1 ? Wk : Wv)) + h * DH * DH;
    v4f wv = *(const v4f*)(Wm + e * DH + d4 * 4);
#pragma unroll
    for (int i = 0; i < 4; ++i) wT_lds[mat][d4 * 4 + i][e] = tobf(wv[i]);
  }
  __syncthreads();

  const int w    = tid >> 5;
  const int lane = tid & 31;
  const int lh   = lane >> 4;
  const int ln   = lane & 15;

  // ---------------- Phase Q (row-major output, scaled) --------------------
#pragma unroll
  for (int i = 0; i < 2; ++i) {
    int t = w * 2 + i;                      // 16 tiles over 8 waves
    int mi = t >> 2, ni = t & 3;
    v8f acc = proj_tile(x_lds, wT_lds[0], mi, ni, lane);
    float bias = bq[h * DH + ni * 16 + ln];
#pragma unroll
    for (int r = 0; r < 8; ++r)
      o_lds[mi * 16 + r + 8 * lh][ni * 16 + ln] = tobf((acc[r] + bias) * 0.125f);
  }
  __syncthreads();
  {
    __bf16* qdst = qkvT + (long)(0 * BH + bh) * S_ * DH;
#pragma unroll
    for (int u2 = 0; u2 < 2; ++u2) {
      int u = u2 * 256 + tid;               // 512 16B units
      int row = u >> 3, c8 = (u & 7) * 8;
      *(v8bf*)(qdst + (long)(s0 + row) * DH + c8) = *(const v8bf*)&o_lds[row][c8];
    }
  }

  // ---------------- Phase K (transposed output, free from C-layout) -------
#pragma unroll
  for (int i = 0; i < 2; ++i) {
    int t = w * 2 + i;
    int mi = t >> 2, ni = t & 3;
    v8f acc = proj_tile(x_lds, wT_lds[1], mi, ni, lane);
    int e = ni * 16 + ln;
    float bias = bk[h * DH + e];
    v8bf o;
#pragma unroll
    for (int r = 0; r < 8; ++r) o[r] = tobf(acc[r] + bias);
    long idx = ((long)(1 * BH + bh) * DH + e) * S_ + s0 + mi * 16 + 8 * lh;
    *(v8bf*)(qkvT + idx) = o;
  }

  // ---------------- Phase V (row-major output) -----------------------------
  __syncthreads();                           // o_lds reads (Q store) done
#pragma unroll
  for (int i = 0; i < 2; ++i) {
    int t = w * 2 + i;
    int mi = t >> 2, ni = t & 3;
    v8f acc = proj_tile(x_lds, wT_lds[2], mi, ni, lane);
    float bias = bv[h * DH + ni * 16 + ln];
#pragma unroll
    for (int r = 0; r < 8; ++r)
      o_lds[mi * 16 + r + 8 * lh][ni * 16 + ln] = tobf(acc[r] + bias);
  }
  __syncthreads();
  {
    __bf16* vdst = qkvT + (long)(2 * BH + bh) * S_ * DH;
#pragma unroll
    for (int u2 = 0; u2 < 2; ++u2) {
      int u = u2 * 256 + tid;
      int row = u >> 3, c8 = (u & 7) * 8;
      *(v8bf*)(vdst + (long)(s0 + row) * DH + c8) = *(const v8bf*)&o_lds[row][c8];
    }
  }
}

// ---------------------------------------------------------------------------
// Kernel 2: flash attention. Block = 128 q rows (8 waves x 16), KV tiles of 32.
// K/V tiles double-buffered via async global->LDS copies (ASYNCcnt).
// Per wave per tile: 4 score WMMAs + 4 PV WMMAs, online softmax in fp32.
// ---------------------------------------------------------------------------
__global__ __launch_bounds__(256) void flash_attn_wmma(
    const __bf16* __restrict__ qkvT, float* __restrict__ out)
{
  __shared__ __align__(16) __bf16 q_lds[128][72];       // [s][e] row-major
  __shared__ __align__(16) __bf16 kT_lds[2][64][40];    // [buf][e][kcol]
  __shared__ __align__(16) __bf16 v_lds[2][32][72];     // [buf][kcol][e]
  __shared__ __align__(16) __bf16 p_lds[8][16][40];     // per-wave P [m][kcol]

  const int tid  = threadIdx.x;
  const int bh   = blockIdx.x;
  const int b    = bh / H_;
  const int h    = bh % H_;
  const int q0   = blockIdx.y * 128;
  const int w    = tid >> 5;
  const int lane = tid & 31;
  const int lh   = lane >> 4;
  const int ln   = lane & 15;

  const __bf16* qR  = qkvT + (long)(0 * BH + bh) * S_ * DH;  // [s][e]
  const __bf16* kTp = qkvT + (long)(1 * BH + bh) * DH * S_;  // [e][s]
  const __bf16* vR  = qkvT + (long)(2 * BH + bh) * S_ * DH;  // [s][e]

  // per-thread staging coordinates (all straight 16B copies)
  const int ke  = tid >> 2, kc8 = (tid & 3) * 8;   // K tile: 64 x 32
  const int vkc = tid >> 3, vc8 = (tid & 7) * 8;   // V tile: 32 x 64

  // stage Q tile (fully contiguous 16KB region) + first KV tile, async
#pragma unroll
  for (int c = 0; c < 4; ++c) {
    int u = c * 256 + tid;                  // 1024 16B units
    int row = u >> 3, c8 = (u & 7) * 8;
    cp16(qR + (long)(q0 + row) * DH + c8, &q_lds[row][c8]);
  }
  cp16(kTp + (long)ke * S_ + 0 + kc8, &kT_lds[0][ke][kc8]);
  cp16(vR + (long)(0 + vkc) * DH + vc8, &v_lds[0][vkc][vc8]);
  async_wait0();
  __syncthreads();

  // Q A-operands: constant across the whole KV loop (kept in VGPRs)
  const int mrow = w * 16 + ln;
  v16bf aq0 = cat16(*(const v8bf*)&q_lds[mrow][0 + 8 * lh],
                    *(const v8bf*)&q_lds[mrow][16 + 8 * lh]);
  v16bf aq1 = cat16(*(const v8bf*)&q_lds[mrow][32 + 8 * lh],
                    *(const v8bf*)&q_lds[mrow][48 + 8 * lh]);

  v8f acco[4] = {{}, {}, {}, {}};
  float rmax[8], rsum[8];
#pragma unroll
  for (int r = 0; r < 8; ++r) { rmax[r] = -1e30f; rsum[r] = 0.0f; }

  for (int it = 0; it < S_ / 32; ++it) {
    const int cur = it & 1;
    const int kv  = it * 32;

    // prefetch next KV tile into the other buffer while we compute
    if (it + 1 < S_ / 32) {
      const int nkv = kv + 32;
      cp16(kTp + (long)ke * S_ + nkv + kc8, &kT_lds[cur ^ 1][ke][kc8]);
      cp16(vR + (long)(nkv + vkc) * DH + vc8, &v_lds[cur ^ 1][vkc][vc8]);
    }

    // scores: two 16(q)x16(k) tiles, K-dim = 64 (e) in two 32-chunks
    v8f s0v = {}, s1v = {};
#pragma unroll
    for (int nt = 0; nt < 2; ++nt) {
      v8f sc = {};
      v16bf bk0;
#pragma unroll
      for (int s = 0; s < 16; ++s) bk0[s] = kT_lds[cur][(s & 7) + 8 * lh + 16 * (s >> 3)][nt * 16 + ln];
      sc = __builtin_amdgcn_wmma_f32_16x16x32_bf16(
          false, aq0, false, bk0, (short)0, sc, false, false);
      v16bf bk1;
#pragma unroll
      for (int s = 0; s < 16; ++s) bk1[s] = kT_lds[cur][32 + (s & 7) + 8 * lh + 16 * (s >> 3)][nt * 16 + ln];
      sc = __builtin_amdgcn_wmma_f32_16x16x32_bf16(
          false, aq1, false, bk1, (short)0, sc, false, false);
      if (nt == 0) s0v = sc; else s1v = sc;
    }

    // online softmax (q was pre-scaled by 1/sqrt(Dh))
    float p0[8], p1[8];
#pragma unroll
    for (int r = 0; r < 8; ++r) {
      float mnew = fmaxf(rmax[r], redmax16(fmaxf(s0v[r], s1v[r])));
      float fac  = __expf(rmax[r] - mnew);
      float e0   = __expf(s0v[r] - mnew);
      float e1   = __expf(s1v[r] - mnew);
      rsum[r] = rsum[r] * fac + redsum16(e0 + e1);
      rmax[r] = mnew;
      p0[r] = e0; p1[r] = e1;
#pragma unroll
      for (int et = 0; et < 4; ++et) acco[et][r] *= fac;
    }

    // P (C-layout) -> per-wave LDS tile -> A-layout
#pragma unroll
    for (int r = 0; r < 8; ++r) {
      p_lds[w][r + 8 * lh][ln]      = tobf(p0[r]);
      p_lds[w][r + 8 * lh][16 + ln] = tobf(p1[r]);
    }
    v16bf ap = cat16(*(const v8bf*)&p_lds[w][ln][8 * lh],
                     *(const v8bf*)&p_lds[w][ln][16 + 8 * lh]);

    // PV: out[16 q][64 e] += P[16][32] x V[32][64]
#pragma unroll
    for (int et = 0; et < 4; ++et) {
      v16bf bv_;
#pragma unroll
      for (int s = 0; s < 16; ++s) bv_[s] = v_lds[cur][(s & 7) + 8 * lh + 16 * (s >> 3)][et * 16 + ln];
      acco[et] = __builtin_amdgcn_wmma_f32_16x16x32_bf16(
          false, ap, false, bv_, (short)0, acco[et], false, false);
    }

    // make the prefetched buffer valid for everyone before next iteration
    async_wait0();
    __syncthreads();
  }

  // normalize + store (contiguous 64B per 16-lane group per VGPR)
  const int srow = q0 + w * 16;
#pragma unroll
  for (int et = 0; et < 4; ++et) {
#pragma unroll
    for (int r = 0; r < 8; ++r) {
      int m = srow + r + 8 * lh;
      int e = et * 16 + ln;
      out[((long)b * S_ + m) * D_ + h * DH + e] = acco[et][r] / rsum[r];
    }
  }
}

// ---------------------------------------------------------------------------
extern "C" void kernel_launch(void* const* d_in, const int* in_sizes, int n_in,
                              void* d_out, int out_size, void* d_ws, size_t ws_size,
                              hipStream_t stream) {
  const float* tokens = (const float*)d_in[0];
  const float* Wq = (const float*)d_in[1];
  const float* bq = (const float*)d_in[2];
  const float* Wk = (const float*)d_in[3];
  const float* bk = (const float*)d_in[4];
  const float* Wv = (const float*)d_in[5];
  const float* bv = (const float*)d_in[6];
  float* out = (float*)d_out;
  __bf16* ws = (__bf16*)d_ws;   // 3 * 192 * 64 * 1024 bf16 = 72 MB

  qkv_proj_wmma<<<dim3(BH, S_ / 64), 256, 0, stream>>>(
      tokens, Wq, bq, Wk, bk, Wv, bv, ws);
  flash_attn_wmma<<<dim3(BH, S_ / 128), 256, 0, stream>>>(ws, out);
}
